// SelfAttention_8108898255232
// MI455X (gfx1250) — matrix operand order repaired and checked by reading, not compiled, not run on hardware
//
#include <hip/hip_runtime.h>
#include <hip/hip_bf16.h>

// ---------- types ----------
typedef __attribute__((ext_vector_type(4)))  float  f32x4;
typedef __attribute__((ext_vector_type(8)))  float  v8f;
typedef __attribute__((ext_vector_type(8)))  __bf16 bf16x8;
typedef __attribute__((ext_vector_type(16))) __bf16 v16bf;

#define D_EMBED 1024
#define SEQ     2048
#define BATCH   8

// ---------- WMMA wrapper ----------
__device__ inline v8f wmma_bf16(v16bf a, v16bf b, v8f c) {
  // D = A(16x32 bf16) * B(32x16 bf16) + C(16x16 f32)
  return __builtin_amdgcn_wmma_f32_16x16x32_bf16(
      /*neg_a=*/false, a, /*neg_b=*/false, b,
      /*c_mod=*/(short)0, c, /*reuse_a=*/false, /*reuse_b=*/false);
}

// ---------- fragment loaders (contiguous k order, wave32) ----------
// A (16x32, 16-bit): lane l holds row M=l&15; elems 0..15 -> K = 16*(l>>4)+e.
__device__ inline v16bf a_frag_from_bf16(const __bf16* __restrict__ M, int ld,
                                         int m0, int k0, int lane) {
  const __bf16* p = M + (size_t)(m0 + (lane & 15)) * ld + (k0 + ((lane >> 4) << 4));
  bf16x8 lo = *(const bf16x8*)(p);        // b128
  bf16x8 hi = *(const bf16x8*)(p + 8);    // b128
  v16bf a;
#pragma unroll
  for (int i = 0; i < 8; ++i) { a[i] = lo[i]; a[8 + i] = hi[i]; }
  return a;
}

// B (32x16, 16-bit): lane l holds column N=l&15; elems 0..15 -> K = 16*(l>>4)+e.
// Source row-major with the B *column* contiguous (X*W^T, Q*K^T).
__device__ inline v16bf b_frag_from_bf16_rows(const __bf16* __restrict__ M, int ld,
                                              int n0, int k0, int lane) {
  const __bf16* p = M + (size_t)(n0 + (lane & 15)) * ld + (k0 + ((lane >> 4) << 4));
  return *(const v16bf*)(p);              // 32B -> 2x b128
}

// B fragment from an LDS-staged K-major tile [32 keys][32 cols] (bf16).
__device__ inline v16bf b_frag_from_lds(const __bf16* tile, int nn, int lane) {
  const __bf16* p = tile + (((lane >> 4) << 4) * 32) + nn * 16 + (lane & 15);
  v16bf b;
#pragma unroll
  for (int e = 0; e < 16; ++e) b[e] = p[e * 32];   // ds_load_u16
  return b;
}

// ---------- Kernel 0: f32 -> bf16 pre-convert (memory bound) ----------
__global__ void __launch_bounds__(256) cvt_bf16_kernel(
    const float* __restrict__ src, __bf16* __restrict__ dst) {
  size_t i = ((size_t)blockIdx.x * 256 + threadIdx.x) * 8;
  f32x4 a = *(const f32x4*)(src + i);
  f32x4 b = *(const f32x4*)(src + i + 4);
  bf16x8 o;
#pragma unroll
  for (int j = 0; j < 4; ++j) { o[j] = (__bf16)a[j]; o[4 + j] = (__bf16)b[j]; }
  *(bf16x8*)(dst + i) = o;
}

// ---------- Kernel 1: fused Q/K/V projection (xb @ Wb^T + b) ----------
// 32x32 per wave (2x2 WMMA tiles), 8 waves per block; pure-bf16 inner loop.
__global__ void __launch_bounds__(256) qkv_kernel(
    const __bf16* __restrict__ xb, const __bf16* __restrict__ Wb,
    const float* __restrict__ bq, const float* __restrict__ bk,
    const float* __restrict__ bv,
    __bf16* __restrict__ q, __bf16* __restrict__ k, __bf16* __restrict__ v) {
  const int lane = threadIdx.x & 31;
  const int wid  = threadIdx.x >> 5;
  const int TILES_PER_MAT = (BATCH * SEQ / 32) * (D_EMBED / 32);   // 512*32 = 16384
  int tile  = blockIdx.x * 8 + wid;                                // 0 .. 49151
  int which = tile >> 14;                                          // /16384
  int t     = tile & (TILES_PER_MAT - 1);
  int m0 = (t >> 5) << 5;
  int n0 = (t & 31) << 5;

  const __bf16* W = Wb + (size_t)which * D_EMBED * D_EMBED;
  const float* bias = (which == 0) ? bq : (which == 1) ? bk : bv;
  __bf16*      out  = (which == 0) ? q  : (which == 1) ? k  : v;

  v8f acc[2][2];
#pragma unroll
  for (int i = 0; i < 2; ++i)
#pragma unroll
    for (int j = 0; j < 2; ++j) acc[i][j] = v8f{0.f,0.f,0.f,0.f,0.f,0.f,0.f,0.f};

#pragma unroll 2
  for (int k0 = 0; k0 < D_EMBED; k0 += 32) {
    v16bf a0 = a_frag_from_bf16(xb, D_EMBED, m0,      k0, lane);
    v16bf a1 = a_frag_from_bf16(xb, D_EMBED, m0 + 16, k0, lane);
    v16bf b0 = b_frag_from_bf16_rows(W, D_EMBED, n0,      k0, lane);
    v16bf b1 = b_frag_from_bf16_rows(W, D_EMBED, n0 + 16, k0, lane);
    acc[0][0] = wmma_bf16(a0, b0, acc[0][0]);
    acc[0][1] = wmma_bf16(a0, b1, acc[0][1]);
    acc[1][0] = wmma_bf16(a1, b0, acc[1][0]);
    acc[1][1] = wmma_bf16(a1, b1, acc[1][1]);
  }
  // C/D layout: lane l, vgpr r -> (M = r + 8*(l>>4), N = l&15)
#pragma unroll
  for (int mi = 0; mi < 2; ++mi) {
    int rbase = m0 + mi * 16 + ((lane >> 4) << 3);
#pragma unroll
    for (int ni = 0; ni < 2; ++ni) {
      int col  = n0 + ni * 16 + (lane & 15);
      float bc = bias[col];
#pragma unroll
      for (int r = 0; r < 8; ++r)
        out[(size_t)(rbase + r) * D_EMBED + col] = (__bf16)(acc[mi][ni][r] + bc);
    }
  }
}

// ---------- Kernel 2: scores = q @ k^T, all batches, bf16 out ----------
// 64x64 per wave (4x4 WMMA tiles): 0.5 fragment loads per WMMA.
__global__ void __launch_bounds__(256) scores_kernel(
    const __bf16* __restrict__ q, const __bf16* __restrict__ kmat,
    __bf16* __restrict__ s) {
  const int lane  = threadIdx.x & 31;
  const int wid   = threadIdx.x >> 5;
  const int batch = blockIdx.y;
  int tile = blockIdx.x * 8 + wid;            // 0 .. 1023  (32x32 tiles of 64x64)
  int m0 = (tile >> 5) << 6;
  int n0 = (tile & 31) << 6;
  const __bf16* qb = q    + (size_t)batch * SEQ * D_EMBED;
  const __bf16* kb = kmat + (size_t)batch * SEQ * D_EMBED;
  __bf16*       sb = s    + (size_t)batch * SEQ * SEQ;

  v8f acc[4][4];
#pragma unroll
  for (int i = 0; i < 4; ++i)
#pragma unroll
    for (int j = 0; j < 4; ++j) acc[i][j] = v8f{0.f,0.f,0.f,0.f,0.f,0.f,0.f,0.f};

#pragma unroll 1
  for (int k0 = 0; k0 < D_EMBED; k0 += 32) {
    v16bf a[4], b[4];
#pragma unroll
    for (int i = 0; i < 4; ++i)
      a[i] = a_frag_from_bf16(qb, D_EMBED, m0 + 16 * i, k0, lane);
#pragma unroll
    for (int j = 0; j < 4; ++j)
      b[j] = b_frag_from_bf16_rows(kb, D_EMBED, n0 + 16 * j, k0, lane);
#pragma unroll
    for (int i = 0; i < 4; ++i)
#pragma unroll
      for (int j = 0; j < 4; ++j)
        acc[i][j] = wmma_bf16(a[i], b[j], acc[i][j]);
  }
#pragma unroll
  for (int mi = 0; mi < 4; ++mi) {
    int rbase = m0 + mi * 16 + ((lane >> 4) << 3);
#pragma unroll
    for (int ni = 0; ni < 4; ++ni) {
      int col = n0 + ni * 16 + (lane & 15);
#pragma unroll
      for (int r = 0; r < 8; ++r)
        sb[(size_t)(rbase + r) * SEQ + col] = (__bf16)acc[mi][ni][r];
    }
  }
}

// ---------- Kernel 3: in-place row softmax of (s + mask) * 1/sqrt(D) ----------
__global__ void __launch_bounds__(256) softmax_kernel(
    __bf16* __restrict__ s, const float* __restrict__ mask) {
  const float scale = 0.03125f;                       // 1/sqrt(1024)
  int batch = blockIdx.y;
  int row   = blockIdx.x;
  int tid   = threadIdx.x;                            // 256 threads, 8 elems each
  int lane = tid & 31, wid = tid >> 5;
  __bf16*      srow = s + ((size_t)batch * SEQ + row) * SEQ;
  const float* mrow = mask + ((size_t)batch * SEQ + row) * SEQ;

  bf16x8 sv = *(const bf16x8*)(srow + tid * 8);
  f32x4 ma  = *(const f32x4*)(mrow + tid * 8);
  f32x4 mb  = *(const f32x4*)(mrow + tid * 8 + 4);
  float t[8];
#pragma unroll
  for (int i = 0; i < 4; ++i) {
    t[i]     = ((float)sv[i]     + ma[i]) * scale;    // mask added BEFORE scale
    t[4 + i] = ((float)sv[4 + i] + mb[i]) * scale;
  }

  __shared__ float red[8];
  float mx = t[0];
#pragma unroll
  for (int i = 1; i < 8; ++i) mx = fmaxf(mx, t[i]);
#pragma unroll
  for (int off = 16; off > 0; off >>= 1) mx = fmaxf(mx, __shfl_xor(mx, off, 32));
  if (lane == 0) red[wid] = mx;
  __syncthreads();
  float rowmax = red[0];
#pragma unroll
  for (int i = 1; i < 8; ++i) rowmax = fmaxf(rowmax, red[i]);
  __syncthreads();

  float e[8]; float sum = 0.f;
#pragma unroll
  for (int i = 0; i < 8; ++i) { e[i] = __expf(t[i] - rowmax); sum += e[i]; }
#pragma unroll
  for (int off = 16; off > 0; off >>= 1) sum += __shfl_xor(sum, off, 32);
  if (lane == 0) red[wid] = sum;
  __syncthreads();
  float rowsum = red[0];
#pragma unroll
  for (int i = 1; i < 8; ++i) rowsum += red[i];
  float inv = 1.0f / rowsum;

  bf16x8 pv;
#pragma unroll
  for (int i = 0; i < 8; ++i) pv[i] = (__bf16)(e[i] * inv);
  *(bf16x8*)(srow + tid * 8) = pv;                    // in-place
}

// ---------- Kernel 4: out = probs @ v ----------
// Block = 256 rows x 32 cols; per 32-key chunk the shared V tile (32x32 bf16)
// is staged to LDS with global_load_async_to_lds_b64 (ASYNCcnt path).
__global__ void __launch_bounds__(256) pv_kernel(
    const __bf16* __restrict__ p, const __bf16* __restrict__ vmat,
    float* __restrict__ out) {
  const int lane  = threadIdx.x & 31;
  const int wid   = threadIdx.x >> 5;
  const int tid   = threadIdx.x;
  const int batch = blockIdx.y;
  int m_base = (blockIdx.x >> 5) << 8;       // *256
  int n0     = (blockIdx.x & 31) << 5;       // *32
  int m0     = m_base + wid * 32;

  const __bf16* pb = p    + (size_t)batch * SEQ * SEQ;
  const __bf16* vb = vmat + (size_t)batch * SEQ * D_EMBED;
  float*        ob = out  + (size_t)batch * SEQ * D_EMBED;

  __shared__ __bf16 vtile[32 * 32];          // [key][col], 2 KB
  unsigned lds_base = (unsigned)(size_t)(&vtile[0]);
  // each of 256 threads copies 8 bytes: key = tid>>3, col chunk = (tid&7)*4
  unsigned lds_dst = lds_base + (unsigned)(((tid >> 3) * 32 + (tid & 7) * 4) * 2);
  const __bf16* vsrc0 = vb + (size_t)(tid >> 3) * D_EMBED + n0 + (tid & 7) * 4;

  v8f acc[2][2];
#pragma unroll
  for (int i = 0; i < 2; ++i)
#pragma unroll
    for (int j = 0; j < 2; ++j) acc[i][j] = v8f{0.f,0.f,0.f,0.f,0.f,0.f,0.f,0.f};

  for (int j0 = 0; j0 < SEQ; j0 += 32) {
    __syncthreads();                         // previous tile fully consumed
    const __bf16* vsrc = vsrc0 + (size_t)j0 * D_EMBED;
    asm volatile("global_load_async_to_lds_b64 %0, %1, off"
                 :: "v"(lds_dst), "v"(vsrc) : "memory");
    asm volatile("s_wait_asynccnt 0" ::: "memory");
    __syncthreads();                         // tile visible to all waves

    v16bf a0 = a_frag_from_bf16(pb, SEQ, m0,      j0, lane);
    v16bf a1 = a_frag_from_bf16(pb, SEQ, m0 + 16, j0, lane);
    v16bf b0 = b_frag_from_lds(vtile, 0, lane);
    v16bf b1 = b_frag_from_lds(vtile, 1, lane);
    acc[0][0] = wmma_bf16(a0, b0, acc[0][0]);
    acc[0][1] = wmma_bf16(a0, b1, acc[0][1]);
    acc[1][0] = wmma_bf16(a1, b0, acc[1][0]);
    acc[1][1] = wmma_bf16(a1, b1, acc[1][1]);
  }

#pragma unroll
  for (int mi = 0; mi < 2; ++mi) {
    int rbase = m0 + mi * 16 + ((lane >> 4) << 3);
#pragma unroll
    for (int ni = 0; ni < 2; ++ni) {
      int col = n0 + ni * 16 + (lane & 15);
#pragma unroll
      for (int r = 0; r < 8; ++r)
        ob[(size_t)(rbase + r) * D_EMBED + col] = acc[mi][ni][r];
    }
  }
}

// ---------- host launcher ----------
extern "C" void kernel_launch(void* const* d_in, const int* in_sizes, int n_in,
                              void* d_out, int out_size, void* d_ws, size_t ws_size,
                              hipStream_t stream) {
  (void)in_sizes; (void)n_in; (void)out_size; (void)ws_size;
  const float* x    = (const float*)d_in[0];
  const float* mask = (const float*)d_in[1];
  const float* Wq   = (const float*)d_in[2];
  const float* bq   = (const float*)d_in[3];
  const float* Wk   = (const float*)d_in[4];
  const float* bk   = (const float*)d_in[5];
  const float* Wv   = (const float*)d_in[6];
  const float* bv   = (const float*)d_in[7];
  float* out = (float*)d_out;

  // workspace (160 MB total):
  //   q @ 0, k @ 32M, v @ 64M            (live through the whole pipeline)
  //   xb @ 96M (32MB), Wb @ 128M (6MB)   (dead after K1)
  //   s  @ 96M (64MB)                    (written by K2 -> aliases xb/Wb safely)
  char* ws = (char*)d_ws;
  __bf16* q  = (__bf16*)(ws);
  __bf16* k  = (__bf16*)(ws + (size_t)32 * 1024 * 1024);
  __bf16* v  = (__bf16*)(ws + (size_t)64 * 1024 * 1024);
  __bf16* xb = (__bf16*)(ws + (size_t)96 * 1024 * 1024);
  __bf16* Wb = (__bf16*)(ws + (size_t)128 * 1024 * 1024);
  __bf16* s  = (__bf16*)(ws + (size_t)96 * 1024 * 1024);

  const size_t NX = (size_t)BATCH * SEQ * D_EMBED;   // 16,777,216
  const size_t NW = (size_t)D_EMBED * D_EMBED;       // 1,048,576

  // K0: pre-convert activations + weights to bf16 (2048 elems per block)
  cvt_bf16_kernel<<<NX / 2048, 256, 0, stream>>>(x,  xb);
  cvt_bf16_kernel<<<NW / 2048, 256, 0, stream>>>(Wq, Wb);
  cvt_bf16_kernel<<<NW / 2048, 256, 0, stream>>>(Wk, Wb + NW);
  cvt_bf16_kernel<<<NW / 2048, 256, 0, stream>>>(Wv, Wb + 2 * NW);

  // K1: 49152 tiles (32x32) / 8 waves
  qkv_kernel<<<6144, 256, 0, stream>>>(xb, Wb, bq, bk, bv, q, k, v);
  // K2: per batch 1024 tiles (64x64) / 8 waves = 128 blocks
  scores_kernel<<<dim3(128, BATCH), 256, 0, stream>>>(q, k, s);
  // K3: one row per block
  softmax_kernel<<<dim3(SEQ, BATCH), 256, 0, stream>>>(s, mask);
  // K4: per batch 8 M-blocks x 32 N-tiles = 256 blocks
  pv_kernel<<<dim3(256, BATCH), 256, 0, stream>>>(s, v, out);
}
